// MultiRelationalGraphDiffusion_25709674234636
// MI455X (gfx1250) — matrix operand order repaired and checked by reading, not compiled, not run on hardware
//
#include <hip/hip_runtime.h>

typedef __attribute__((ext_vector_type(16))) _Float16 v16h;
typedef __attribute__((ext_vector_type(8)))  _Float16 v8h;
typedef __attribute__((ext_vector_type(8)))  float    v8f;

#define NB   8
#define NN   2048
#define NF   256
#define LDT  40          // padded LDS row stride in halves (conflict-free for b128)

__device__ __forceinline__ v16h concat8(v8h lo, v8h hi) {
  v16h r;
#pragma unroll
  for (int i = 0; i < 8; ++i) { r[i] = lo[i]; r[8 + i] = hi[i]; }
  return r;
}

// ---------------------------------------------------------------------------
// 1) inv row-sum of adjacency: one wave32 per row
// ---------------------------------------------------------------------------
__global__ __launch_bounds__(256) void rowsum_kernel(
    const float* __restrict__ adj, float* __restrict__ inv_rs) {
  const int lane = threadIdx.x & 31;
  const int row  = blockIdx.x * 8 + (threadIdx.x >> 5);  // global row in [0, B*N)
  const float4* p = (const float4*)(adj + (size_t)row * NN);
  float s = 0.f;
#pragma unroll
  for (int i = 0; i < NN / 128; ++i) {
    float4 f = p[lane + i * 32];
    s += f.x + f.y + f.z + f.w;
  }
#pragma unroll
  for (int off = 16; off > 0; off >>= 1) s += __shfl_xor(s, off, 32);
  if (lane == 0) inv_rs[row] = 1.f / (s + 1e-9f);
}

// ---------------------------------------------------------------------------
// 1b) optional: pre-normalize adjacency to f16 once (adj16 = adj * inv_rs)
// ---------------------------------------------------------------------------
__global__ __launch_bounds__(256) void convert_adj_kernel(
    const float* __restrict__ adj, const float* __restrict__ inv_rs,
    _Float16* __restrict__ adj16) {
  const size_t row = blockIdx.x;            // 0 .. B*N-1
  const float  s   = inv_rs[row];
  const int    c   = threadIdx.x * 8;
  const float4* src = (const float4*)(adj + row * NN + c);
  float4 a = src[0], b = src[1];
  _Float16 t[8];
  t[0] = (_Float16)(a.x * s); t[1] = (_Float16)(a.y * s);
  t[2] = (_Float16)(a.z * s); t[3] = (_Float16)(a.w * s);
  t[4] = (_Float16)(b.x * s); t[5] = (_Float16)(b.y * s);
  t[6] = (_Float16)(b.z * s); t[7] = (_Float16)(b.w * s);
  *(uint4*)(adj16 + row * NN + c) = *(uint4*)t;
}

// ---------------------------------------------------------------------------
// 2) Double-buffered tiled WMMA GEMM:
//      D[m][n] = sum_k A[m][k] * B[n][k]
//    A: f16 (pre-normalized adj16) or f32 (adj row-scaled inline / h);
//    B: f32 (W) or f16 (feature-major curr).  Both B layouts are [N][K].
//    Writes nextT (f16, [b][n][m]) and accumulates outT (f32, [b][n][m]).
// ---------------------------------------------------------------------------
template <bool A_F16, bool A_SCALE, bool B_F16, bool FIRST>
__global__ __launch_bounds__(256) void gemm_diffuse(
    const float*    __restrict__ Af32,    // [b][2048][K]
    const _Float16* __restrict__ A16,     // [b][2048][K] (pre-normalized)
    const float*    __restrict__ inv_rs,  // [b][2048]
    const float*    __restrict__ Bf32,    // [Ntot][K]  (W)
    const _Float16* __restrict__ Bf16,    // [b][Ntot][K] (curr, feature-major)
    _Float16*       __restrict__ nextT,   // [b][256][2048] f16
    float*          __restrict__ outT,    // [b][256][2048] f32
    const float*    __restrict__ conv_w,
    int k_idx, int K) {
  __shared__ _Float16 lA[2][128 * LDT];
  __shared__ _Float16 lB[2][128 * LDT];

  const int b    = blockIdx.z;
  const int m0   = blockIdx.y * 128;
  const int n0   = blockIdx.x * 128;
  const int tid  = threadIdx.x;
  const int lane = tid & 31;
  const int wv   = tid >> 5;
  const int wm   = wv & 1;   // 64-row sub-tile
  const int wn   = wv >> 1;  // 32-col sub-tile

  const float*    Abf = A_F16 ? nullptr : (Af32 + (size_t)b * NN * (size_t)K);
  const _Float16* Abh = A_F16 ? (A16 + (size_t)b * NN * (size_t)K) : nullptr;
  const _Float16* Bs16 = B_F16 ? (Bf16 + (size_t)b * NF * (size_t)K) : nullptr;

  // global->LDS loader mapping: 2 threads per 32-wide row
  const int lr = tid >> 1;          // 0..127
  const int lc = (tid & 1) * 16;    // 0 or 16

  float scaleA = 1.0f;
  if constexpr (A_SCALE) scaleA = inv_rs[b * NN + m0 + lr];

  // register staging for the next tile
  float4 raf[4]; uint4 rau[2];
  float4 rbf[4]; uint4 rbu[2];

  auto loadA = [&](int kk) {
    if constexpr (A_F16) {
      const uint4* src = (const uint4*)(Abh + (size_t)(m0 + lr) * K + kk + lc);
      rau[0] = src[0]; rau[1] = src[1];
    } else {
      const float4* src = (const float4*)(Abf + (size_t)(m0 + lr) * K + kk + lc);
      raf[0] = src[0]; raf[1] = src[1]; raf[2] = src[2]; raf[3] = src[3];
    }
  };
  auto loadB = [&](int kk) {
    if constexpr (B_F16) {
      const uint4* src = (const uint4*)(Bs16 + (size_t)(n0 + lr) * K + kk + lc);
      rbu[0] = src[0]; rbu[1] = src[1];
    } else {
      const float4* src = (const float4*)(Bf32 + (size_t)(n0 + lr) * K + kk + lc);
      rbf[0] = src[0]; rbf[1] = src[1]; rbf[2] = src[2]; rbf[3] = src[3];
    }
  };
  auto storeA = [&](int buf) {
    _Float16* base = &lA[buf][lr * LDT + lc];
    if constexpr (A_F16) {
      ((uint4*)base)[0] = rau[0]; ((uint4*)base)[1] = rau[1];
    } else {
      _Float16 tmp[16];
#pragma unroll
      for (int q = 0; q < 4; ++q) {
        tmp[4 * q + 0] = (_Float16)(raf[q].x * scaleA);
        tmp[4 * q + 1] = (_Float16)(raf[q].y * scaleA);
        tmp[4 * q + 2] = (_Float16)(raf[q].z * scaleA);
        tmp[4 * q + 3] = (_Float16)(raf[q].w * scaleA);
      }
      ((v8h*)base)[0] = *(v8h*)&tmp[0];
      ((v8h*)base)[1] = *(v8h*)&tmp[8];
    }
  };
  auto storeB = [&](int buf) {
    _Float16* base = &lB[buf][lr * LDT + lc];
    if constexpr (B_F16) {
      ((uint4*)base)[0] = rbu[0]; ((uint4*)base)[1] = rbu[1];
    } else {
      _Float16 tmp[16];
#pragma unroll
      for (int q = 0; q < 4; ++q) {
        tmp[4 * q + 0] = (_Float16)rbf[q].x;
        tmp[4 * q + 1] = (_Float16)rbf[q].y;
        tmp[4 * q + 2] = (_Float16)rbf[q].z;
        tmp[4 * q + 3] = (_Float16)rbf[q].w;
      }
      ((v8h*)base)[0] = *(v8h*)&tmp[0];
      ((v8h*)base)[1] = *(v8h*)&tmp[8];
    }
  };

  v8f acc[4][2];
#pragma unroll
  for (int f = 0; f < 4; ++f)
#pragma unroll
    for (int g = 0; g < 2; ++g) acc[f][g] = (v8f){0, 0, 0, 0, 0, 0, 0, 0};

  const int arow = lane & 15;
  const int asel = lane >> 4;       // 0/1: which K-half this lane owns

  // ---- pipeline prologue: tile 0 -> lds[0]
  loadA(0); loadB(0);
  storeA(0); storeB(0);
  __syncthreads();

  int buf = 0;
  for (int kk = 0; kk < K; kk += 32) {
    const bool more = (kk + 32) < K;
    if (more) { loadA(kk + 32); loadB(kk + 32); }  // overlap with WMMA below

    // ---- fragments per ISA VGPR layouts (from lds[buf])
    v16h af[4];
#pragma unroll
    for (int f = 0; f < 4; ++f) {
      const int r = wm * 64 + f * 16 + arow;
      const v8h* p = (const v8h*)&lA[buf][r * LDT];
      af[f] = concat8(p[asel], p[asel + 2]);  // K {0..7,16..23} or {8..15,24..31}
    }
    v16h bfr[2];
#pragma unroll
    for (int g = 0; g < 2; ++g) {
      const int c = wn * 32 + g * 16 + arow;  // N index
      const v8h* p = (const v8h*)&lB[buf][c * LDT];
      bfr[g] = concat8(p[asel], p[asel + 2]); // K {0..7,16..23} or {8..15,24..31}
    }
#pragma unroll
    for (int f = 0; f < 4; ++f)
#pragma unroll
      for (int g = 0; g < 2; ++g)
        acc[f][g] = __builtin_amdgcn_wmma_f32_16x16x32_f16(
            false, af[f], false, bfr[g], (short)0, acc[f][g], false, false);

    if (more) {
      storeA(buf ^ 1); storeB(buf ^ 1);  // other buffer: safe, was drained
      __syncthreads();
      buf ^= 1;
    }
  }

  // ---- epilogue: D lane layout => 8 consecutive M for fixed N (contiguous
  // in feature-major storage)
  const float cw = conv_w[k_idx];
#pragma unroll
  for (int f = 0; f < 4; ++f) {
#pragma unroll
    for (int g = 0; g < 2; ++g) {
      const int Mb = m0 + wm * 64 + f * 16 + 8 * (lane >> 4);
      const int Ng = n0 + wn * 32 + g * 16 + (lane & 15);
      const size_t off = ((size_t)b * NF + Ng) * NN + Mb;
      v8f c = acc[f][g];
      v8h h8;
#pragma unroll
      for (int v = 0; v < 8; ++v) h8[v] = (_Float16)c[v];
      *(v8h*)(nextT + off) = h8;
      float* op = outT + off;
      if constexpr (FIRST) {
        float4 r0 = {cw * c[0], cw * c[1], cw * c[2], cw * c[3]};
        float4 r1 = {cw * c[4], cw * c[5], cw * c[6], cw * c[7]};
        *(float4*)op = r0;
        *(float4*)(op + 4) = r1;
      } else {
        float4 o0 = *(float4*)op;
        float4 o1 = *(float4*)(op + 4);
        o0.x += cw * c[0]; o0.y += cw * c[1]; o0.z += cw * c[2]; o0.w += cw * c[3];
        o1.x += cw * c[4]; o1.y += cw * c[5]; o1.z += cw * c[6]; o1.w += cw * c[7];
        *(float4*)op = o0;
        *(float4*)(op + 4) = o1;
      }
    }
  }
}

// ---------------------------------------------------------------------------
// 3) transpose back to [b][n][o] + bias + LeakyReLU(0.2) via 32x32 LDS tile
// ---------------------------------------------------------------------------
__global__ __launch_bounds__(256) void finalize_kernel(
    const float* __restrict__ outT, const float* __restrict__ conv_b,
    float* __restrict__ out) {
  __shared__ float tl[32][33];
  const int b  = blockIdx.z;
  const int n0 = blockIdx.x * 32;
  const int o0 = blockIdx.y * 32;
  const int t  = threadIdx.x;
  const int r  = t >> 3;         // 0..31
  const int c4 = (t & 7) * 4;    // 0,4,...,28
  const float cb = conv_b[0];

  float4 v = *(const float4*)(outT + ((size_t)b * NF + o0 + r) * NN + n0 + c4);
  tl[r][c4 + 0] = v.x; tl[r][c4 + 1] = v.y; tl[r][c4 + 2] = v.z; tl[r][c4 + 3] = v.w;
  __syncthreads();

  float4 w;
  float x;
  x = tl[c4 + 0][r] + cb; w.x = (x >= 0.f) ? x : 0.2f * x;
  x = tl[c4 + 1][r] + cb; w.y = (x >= 0.f) ? x : 0.2f * x;
  x = tl[c4 + 2][r] + cb; w.z = (x >= 0.f) ? x : 0.2f * x;
  x = tl[c4 + 3][r] + cb; w.w = (x >= 0.f) ? x : 0.2f * x;
  *(float4*)(out + ((size_t)b * NN + n0 + r) * NF + o0 + c4) = w;
}

// ---------------------------------------------------------------------------
extern "C" void kernel_launch(void* const* d_in, const int* in_sizes, int n_in,
                              void* d_out, int out_size, void* d_ws,
                              size_t ws_size, hipStream_t stream) {
  const float* h   = (const float*)d_in[0];   // [8,2048,256]
  const float* adj = (const float*)d_in[1];   // [8,2048,2048]
  const float* W   = (const float*)d_in[2];   // [256,256]  (out,in)
  const float* cw  = (const float*)d_in[3];   // [4]
  const float* cb  = (const float*)d_in[4];   // [1]
  float* out = (float*)d_out;

  char* ws = (char*)d_ws;
  float*    inv_rs = (float*)ws;                               // 64 KB
  _Float16* curr0  = (_Float16*)(ws + (1 << 16));              // 8 MB
  _Float16* curr1  = curr0 + (size_t)NB * NF * NN;             // 8 MB
  float*    outT   = (float*)(curr1 + (size_t)NB * NF * NN);   // 16 MB
  _Float16* adj16  = (_Float16*)(outT + (size_t)NB * NF * NN); // 64 MB (opt)

  const size_t base_bytes  = (size_t)(1 << 16) +
                             2 * (size_t)NB * NF * NN * sizeof(_Float16) +
                             (size_t)NB * NF * NN * sizeof(float);
  const size_t adj16_bytes = (size_t)NB * NN * NN * sizeof(_Float16);
  const bool   big_ws      = ws_size >= base_bytes + adj16_bytes;

  // 1) row sums (one wave per row)
  rowsum_kernel<<<(NB * NN) / 8, 256, 0, stream>>>(adj, inv_rs);

  dim3 g(NF / 128, NN / 128, NB);  // (2, 16, 8)

  // 2) h_trans = h @ W^T  -> curr0 (f16, feature-major); outT = cw[0]*x
  gemm_diffuse<false, false, false, true><<<g, 256, 0, stream>>>(
      h, nullptr, inv_rs, W, nullptr, curr0, outT, cw, 0, NF);

  // 3) three diffusion steps: curr_k = adj_norm @ curr_{k-1}; outT += cw[k]*x
  if (big_ws) {
    // pre-normalize adj once to f16: no per-pass conversion, half the L2 bytes
    convert_adj_kernel<<<NB * NN, 256, 0, stream>>>(adj, inv_rs, adj16);
    gemm_diffuse<true, false, true, false><<<g, 256, 0, stream>>>(
        nullptr, adj16, inv_rs, nullptr, curr0, curr1, outT, cw, 1, NN);
    gemm_diffuse<true, false, true, false><<<g, 256, 0, stream>>>(
        nullptr, adj16, inv_rs, nullptr, curr1, curr0, outT, cw, 2, NN);
    gemm_diffuse<true, false, true, false><<<g, 256, 0, stream>>>(
        nullptr, adj16, inv_rs, nullptr, curr0, curr1, outT, cw, 3, NN);
  } else {
    gemm_diffuse<false, true, true, false><<<g, 256, 0, stream>>>(
        adj, nullptr, inv_rs, nullptr, curr0, curr1, outT, cw, 1, NN);
    gemm_diffuse<false, true, true, false><<<g, 256, 0, stream>>>(
        adj, nullptr, inv_rs, nullptr, curr1, curr0, outT, cw, 2, NN);
    gemm_diffuse<false, true, true, false><<<g, 256, 0, stream>>>(
        adj, nullptr, inv_rs, nullptr, curr0, curr1, outT, cw, 3, NN);
  }

  // 4) transpose + bias + LeakyReLU into d_out [b][n][o]
  dim3 gf(NN / 32, NF / 32, NB);   // (64, 8, 8)
  finalize_kernel<<<gf, 256, 0, stream>>>(outT, cb, out);
}
